// FractalTransformer_26766236189056
// MI455X (gfx1250) — matrix operand order repaired and checked by reading, not compiled, not run on hardware
//
#include <hip/hip_runtime.h>
#include <hip/hip_bf16.h>

// ---------------------------------------------------------------------------
// FractalTransformer fused implementation for gfx1250 (MI455X, wave32, WMMA).
//
// Pipeline (all on `stream`, scratch in d_ws, ~6 MB used):
//   1) X  = x    @ w_pre^T                      (gemm_rt, beta=0)
//   2) q  = X + X@w_q^T ; k = X + X@w_k^T ; v = X + X@w_va^T
//   3) vsum = fused swishmax-attention          (fractal_attn, online reduce)
//   4) t  = vsum + vsum@w_vo^T + X
//   5) out= t @ w_end^T
//
// All matmuls use v_wmma_f32_16x16x32_bf16 (fp32 accumulate). raw/logits
// ([2,512,512,256], 537 MB) are NEVER materialized: they live as 16x256
// LDS tiles inside fractal_attn. bf16 conversion uses native packed cvts
// (__builtin_convertvector), not manual integer RNE.
// ---------------------------------------------------------------------------

typedef __attribute__((ext_vector_type(16))) __bf16   v16bf;
typedef __attribute__((ext_vector_type(8)))  __bf16   v8bf;
typedef __attribute__((ext_vector_type(16))) float    v16f;
typedef __attribute__((ext_vector_type(8)))  float    v8f;
typedef __attribute__((ext_vector_type(4)))  unsigned v4u;

union Frag16 {                 // one 16x16x32 bf16 WMMA operand (8 VGPRs)
    v16bf v;
    v8bf  h8[2];
    v4u   q4[2];
};

#define DD   256   // feature dim
#define SS   512   // sequence length
#define BB   2     // batch

// ---------------------------------------------------------------------------
// Generic residual GEMM: Out[M,256] = A[M,256] @ W[256,256]^T (+A) (+R)
// One 16x16 output tile per wave. B-fragment for A@W^T reads W row-major
// directly (B[k][n] = W[n][k]): lane holds column n=lr, contiguous K.
// ---------------------------------------------------------------------------
__global__ __launch_bounds__(256) void gemm_rt(
    const float* __restrict__ A, const float* __restrict__ W,
    const float* __restrict__ R, float* __restrict__ Out,
    int addA, int addR)
{
    const int lane = threadIdx.x & 31;
    const int wave = threadIdx.x >> 5;
    const int tile = blockIdx.x * 8 + wave;      // 1024 tiles total
    const int mt = tile >> 4;                    // 64 M-tiles (M=1024)
    const int nt = tile & 15;                    // 16 N-tiles
    const int hi = lane >> 4;                    // half-wave select
    const int lr = lane & 15;

    v8f acc = {};
    const float* arow = A + (mt * 16 + lr) * DD;
    const float* wrow = W + (nt * 16 + lr) * DD;
#pragma unroll
    for (int kt = 0; kt < 8; ++kt) {
        // A fragment: K = kt*32 + hi*16 + e contiguous (matches B).
        //             32B-aligned v8f loads.
        Frag16 a, b;
        v8f a0 = *(const v8f*)(arow + kt * 32 + hi * 16);
        v8f a1 = *(const v8f*)(arow + kt * 32 + hi * 16 + 8);
        a.h8[0] = __builtin_convertvector(a0, v8bf);   // packed cvt, RNE
        a.h8[1] = __builtin_convertvector(a1, v8bf);
        // B fragment: lane = column n, K = kt*32 + hi*16 + e contiguous in W row
        v16f bw = *(const v16f*)(wrow + kt * 32 + hi * 16);
        b.v = __builtin_convertvector(bw, v16bf);
        acc = __builtin_amdgcn_wmma_f32_16x16x32_bf16(
                  false, a.v, false, b.v, (short)0, acc, false, false);
    }
    // C layout: acc[r] <-> row M = hi*8 + r, col N = lr
#pragma unroll
    for (int r = 0; r < 8; ++r) {
        int row = mt * 16 + hi * 8 + r;
        int col = nt * 16 + lr;
        float o = acc[r];
        if (addA) o += A[row * DD + col];
        if (addR) o += R[row * DD + col];
        Out[row * DD + col] = o;
    }
}

// ---------------------------------------------------------------------------
// Fused FractalAttention core.
// Grid: (128, 2) = (sq-tiles of 4, batch). Block: 512 threads = 16 waves.
// Wave w owns d-columns [16w, 16w+16); its 8 w_al B-fragments stay in VGPRs.
// For each sk-tile (16 rows): build raw = k_tile * q_sq in LDS (bf16 for
// WMMA A, k kept fp32 for the residual term), do 8 WMMAs -> logits tile,
// then online-swishmax update per d column (reduce over M via lane^16).
//   vsum[d] = sum_sk v*x*exp(x-m) / (sum_sk |x*exp(x-m)| + 1)
// ---------------------------------------------------------------------------
#define SQT 4
#define KFP (DD + 8)    // kf row stride (floats):  264 -> 32B aligned, bank-spread
#define RBP (DD + 8)    // rawb row stride (halfs): 264 -> 16B aligned, bank-spread

__global__ __launch_bounds__(512, 1) void fractal_attn(
    const float* __restrict__ q, const float* __restrict__ k,
    const float* __restrict__ v, const float* __restrict__ w_al,
    float* __restrict__ vsum_out)
{
    __shared__ __align__(32) float  qs[SQT][DD];        // 4 KB
    __shared__ __align__(32) float  kf[16][KFP];        // 16.9 KB
    __shared__ __align__(16) __bf16 rawb[SQT][16][RBP]; // 33.8 KB

    const int t     = threadIdx.x;
    const int lane  = t & 31;
    const int wave  = t >> 5;          // 0..15 -> N-tile (d block)
    const int hi    = lane >> 4;
    const int lr    = lane & 15;
    const int b     = blockIdx.y;
    const int sqb   = blockIdx.x * SQT;
    const int d     = wave * 16 + lr;  // this lane's feature column

    // stage q vectors for the 4 sq columns
    for (int i = t; i < SQT * DD; i += 512) {
        int s = i >> 8, c = i & 255;
        qs[s][c] = q[((b * SS) + sqb + s) * DD + c];
    }

    // Preload this wave's w_al B-fragments (persistent across whole sk loop).
    Frag16 bfrag[8];
    {
        const float* wrow = w_al + (wave * 16 + lr) * DD;
#pragma unroll
        for (int kt = 0; kt < 8; ++kt) {
            v16f bw = *(const v16f*)(wrow + kt * 32 + hi * 16);
            bfrag[kt].v = __builtin_convertvector(bw, v16bf);
        }
    }

    float m[SQT], num[SQT], den[SQT];
#pragma unroll
    for (int s = 0; s < SQT; ++s) { m[s] = -__builtin_inff(); num[s] = 0.f; den[s] = 0.f; }

    const float* kbase = k + b * SS * DD;
    const float* vbase = v + b * SS * DD;

    for (int mt = 0; mt < SS / 16; ++mt) {
        __syncthreads();   // previous tile fully consumed (also fences qs on mt=0)

        // Cooperative build: k tile (fp32) + raw tiles (bf16) for all 4 sq.
        // Per thread: one 32B global load, one 32B LDS store, 4x(v8f mul +
        // packed cvt + 16B LDS store). All vector, all aligned.
        {
            int idx = t * 8;            // 512 thr * 8 = 4096 = 16*256
            int row = idx >> 8;
            int col = idx & 255;
            v8f kv = *(const v8f*)(kbase + (mt * 16 + row) * DD + col);
            *(v8f*)&kf[row][col] = kv;
#pragma unroll
            for (int s = 0; s < SQT; ++s) {
                v8f pr = kv * *(const v8f*)&qs[s][col];
                *(v8bf*)&rawb[s][row][col] = __builtin_convertvector(pr, v8bf);
            }
        }
        __syncthreads();

        // Per-lane invariants for this sk-tile
        float lv[8], kres[8];
#pragma unroll
        for (int r = 0; r < 8; ++r) {
            int row = hi * 8 + r;
            lv[r]   = vbase[(mt * 16 + row) * DD + d];
            kres[r] = kf[row][d];
        }

#pragma unroll
        for (int s = 0; s < SQT; ++s) {
            // logits tile = raw @ w_al^T  (fp32 accum); A frags via 2x ds_load_b128
            v8f acc = {};
#pragma unroll
            for (int kt = 0; kt < 8; ++kt) {
                Frag16 a;
                const __bf16* ar = &rawb[s][lr][kt * 32 + hi * 16];
                a.q4[0] = *(const v4u*)(ar);        // halves K=kt*32+hi*16+0..7
                a.q4[1] = *(const v4u*)(ar + 8);    // halves K=kt*32+hi*16+8..15
                acc = __builtin_amdgcn_wmma_f32_16x16x32_bf16(
                          false, a.v, false, bfrag[kt].v, (short)0, acc, false, false);
            }
            const float qsd = qs[s][d];

            // residual + per-lane 8-row stats
            float lg[8];
            float lmax = -__builtin_inff();
#pragma unroll
            for (int r = 0; r < 8; ++r) {
                float x = acc[r] + kres[r] * qsd;   // logits = raw@W^T + raw
                lg[r] = x;
                lmax  = fmaxf(lmax, x);
            }
            // merge the other 8 rows (lane ^ 16 holds same d, rows +/-8)
            lmax = fmaxf(lmax, __shfl_xor(lmax, 16, 32));

            // online rescale of running sums
            float nm = fmaxf(m[s], lmax);
            float rs = __expf(m[s] - nm);           // exp(-inf)=0 on first tile
            num[s] *= rs; den[s] *= rs; m[s] = nm;

            float pn = 0.f, pd = 0.f;
#pragma unroll
            for (int r = 0; r < 8; ++r) {
                float e = lg[r] * __expf(lg[r] - nm);  // x*exp(x-m)
                pn += lv[r] * e;
                pd += fabsf(e);
            }
            pn += __shfl_xor(pn, 16, 32);
            pd += __shfl_xor(pd, 16, 32);
            num[s] += pn; den[s] += pd;
        }
    }

    if (hi == 0) {
#pragma unroll
        for (int s = 0; s < SQT; ++s)
            vsum_out[((b * SS) + sqb + s) * DD + d] = num[s] / (den[s] + 1.0f);
    }
}

// ---------------------------------------------------------------------------
extern "C" void kernel_launch(void* const* d_in, const int* in_sizes, int n_in,
                              void* d_out, int out_size, void* d_ws, size_t ws_size,
                              hipStream_t stream)
{
    const float* x     = (const float*)d_in[0];
    const float* w_pre = (const float*)d_in[1];
    const float* w_q   = (const float*)d_in[2];
    const float* w_k   = (const float*)d_in[3];
    const float* w_va  = (const float*)d_in[4];
    const float* w_al  = (const float*)d_in[5];
    const float* w_vo  = (const float*)d_in[6];
    const float* w_end = (const float*)d_in[7];
    float* out = (float*)d_out;

    const size_t N = (size_t)BB * SS * DD;   // 262144 floats per tensor
    float* X    = (float*)d_ws;
    float* qb   = X    + N;
    float* kb   = qb   + N;
    float* vb   = kb   + N;
    float* vsum = vb   + N;
    float* tb   = vsum + N;                  // total 6 MB scratch

    dim3 gblk(256), ggrd(128);               // 1024 tiles / 8 waves per block

    // 1) X = x @ w_pre^T
    gemm_rt<<<ggrd, gblk, 0, stream>>>(x, w_pre, nullptr, X, 0, 0);
    // 2) residual projections
    gemm_rt<<<ggrd, gblk, 0, stream>>>(X, w_q,  nullptr, qb, 1, 0);
    gemm_rt<<<ggrd, gblk, 0, stream>>>(X, w_k,  nullptr, kb, 1, 0);
    gemm_rt<<<ggrd, gblk, 0, stream>>>(X, w_va, nullptr, vb, 1, 0);
    // 3) fused attention + swishmax (the 68.7 GFLOP core)
    fractal_attn<<<dim3(SS / SQT, BB), 512, 0, stream>>>(qb, kb, vb, w_al, vsum);
    // 4) t = vsum + vsum@w_vo^T + X
    gemm_rt<<<ggrd, gblk, 0, stream>>>(vsum, w_vo, X, tb, 1, 1);
    // 5) out = t @ w_end^T
    gemm_rt<<<ggrd, gblk, 0, stream>>>(tb, w_end, nullptr, out, 0, 0);
}
